// MultiheadAttentionSIM_46265387712829
// MI455X (gfx1250) — matrix operand order repaired and checked by reading, not compiled, not run on hardware
//
#include <hip/hip_runtime.h>

typedef __attribute__((ext_vector_type(16))) _Float16 v16h;
typedef __attribute__((ext_vector_type(8)))  float    v8f;
typedef __attribute__((ext_vector_type(4)))  float    v4f;

#define B_SZ   2
#define S_LEN  2048
#define D_DIM  1024
#define NH     16
#define HD     64
#define SSTRIDE 2080   // 2048 + padding to dodge LDS bank conflicts

union FragH {
  v16h v;
  v4f  q[2];
};

// ---------------------------------------------------------------------------
// A-fragment: 16x32 f16 (M x K), row-major source with leading dim `lda`
// (halves). Wave32 layout per CDNA5 ISA 7.12.2:
//   lane<16 : row M=lane,   V0-3 <- K=0..7,  V4-7 <- K=16..23
//   lane>=16: row M=lane-16,V0-3 <- K=8..15, V4-7 <- K=24..31
// ---------------------------------------------------------------------------
__device__ inline v16h load_a_frag(const _Float16* __restrict__ src, int lda) {
  const int lane  = threadIdx.x & 31;
  const int row   = lane & 15;
  const int khalf = (lane >> 4) << 3;           // 0 or 8
  FragH f;
  const _Float16* p = src + (size_t)row * lda + khalf;
  f.q[0] = *(const v4f*)(p);                    // K = khalf .. khalf+7
  f.q[1] = *(const v4f*)(p + 16);               // K = 16+khalf .. 16+khalf+7
  return f.v;
}

// ---------------------------------------------------------------------------
// B-fragment: 32x16 f16 (K x N) where element(k,n) = src[n*ldb + k]
// (i.e. source rows are the N columns — contiguous in K).
//   lane<16 : col N=lane,    V0-3 <- K=0..7,  V4-7 <- K=16..23
//   lane>=16: col N=lane-16, V0-3 <- K=8..15, V4-7 <- K=24..31
// ---------------------------------------------------------------------------
__device__ inline v16h load_b_frag(const _Float16* __restrict__ src, int ldb) {
  const int lane = threadIdx.x & 31;
  const int n    = lane & 15;
  const int koff = (lane >> 4) << 3;            // 0 or 8
  FragH f;
  const _Float16* p = src + (size_t)n * ldb + koff;
  f.q[0] = *(const v4f*)(p);                    // K = koff .. koff+7
  f.q[1] = *(const v4f*)(p + 16);               // K = 16+koff .. 16+koff+7
  return f.v;
}

// ---------------------------------------------------------------------------
// f32 -> f16 bulk convert, 4 elements per thread (n is a multiple of 4)
// ---------------------------------------------------------------------------
__global__ __launch_bounds__(256) void cvt_kernel(const float* __restrict__ in,
                                                  _Float16* __restrict__ out,
                                                  int n) {
  int i = (blockIdx.x * 256 + threadIdx.x) * 4;
  if (i + 3 < n) {
    v4f v = *(const v4f*)(in + i);
    out[i + 0] = (_Float16)v.x;
    out[i + 1] = (_Float16)v.y;
    out[i + 2] = (_Float16)v.z;
    out[i + 3] = (_Float16)v.w;
  }
}

// ---------------------------------------------------------------------------
// Projection GEMM: out[n,j] = (sum_d X[n,d]*W[j,d] + bias[j]) * scale
// stored f16 into [B, NH, S, HD] layout (n = b*S+s, j = h*HD+hd).
// grid = (M/16, N/128), block = 256 (8 waves); each wave -> one 16x16 tile.
// ---------------------------------------------------------------------------
__global__ __launch_bounds__(256) void proj_kernel(
    const _Float16* __restrict__ X16,   // [B*S, D]
    const _Float16* __restrict__ W16,   // [D, D] row-major (out_feat x in_feat)
    const float*    __restrict__ bias,  // [D]
    _Float16*       __restrict__ out,   // [B, NH, S, HD]
    float scale)
{
  const int wave    = threadIdx.x >> 5;
  const int lane    = threadIdx.x & 31;
  const int col     = lane & 15;
  const int rowbase = (lane >> 4) << 3;
  const int mbase   = blockIdx.x * 16;
  const int jbase   = blockIdx.y * 128 + wave * 16;

  const _Float16* arow = X16 + (size_t)mbase * D_DIM;
  const _Float16* brow = W16 + (size_t)jbase * D_DIM;

  v8f c = {};
  for (int k0 = 0; k0 < D_DIM; k0 += 32) {
    v16h a  = load_a_frag(arow + k0, D_DIM);
    v16h bf = load_b_frag(brow + k0, D_DIM);
    c = __builtin_amdgcn_wmma_f32_16x16x32_f16(false, a, false, bf,
                                               (short)0, c, false, false);
  }

  const int   j  = jbase + col;
  const float bv = bias[j];
  const int   h  = j >> 6;      // j / HD
  const int   hd = j & 63;      // j % HD
#pragma unroll
  for (int r = 0; r < 8; ++r) {
    const int n  = mbase + rowbase + r;
    const int bb = n >> 11;     // n / S_LEN
    const int s  = n & 2047;    // n % S_LEN
    const float v = (c[r] + bv) * scale;
    out[(((size_t)bb * NH + h) * S_LEN + s) * HD + hd] = (_Float16)v;
  }
}

// ---------------------------------------------------------------------------
// Attention: per workgroup, one (b, 16-query tile). For each head:
//  - 8 waves compute 16x16 score tiles via chained WMMA over HD=64
//  - scores staged in LDS [16][SSTRIDE] f32
//  - softmax over 2048 keys (16 threads per row)
//  - head mean accumulated in a second LDS buffer; global out written ONCE
//    (eliminates the per-head global read-modify-write -> ~1 GB HBM saved)
// LDS: 2 * 16*SSTRIDE f32 + reductions = ~267 KB (<= 320 KB/WGP on CDNA5)
// ---------------------------------------------------------------------------
__global__ __launch_bounds__(256) void attn_kernel(
    const _Float16* __restrict__ qbuf,  // [B, NH, S, HD], pre-scaled by 1/8
    const _Float16* __restrict__ kbuf,  // [B, NH, S, HD]
    float*          __restrict__ out)   // [B, S, S]
{
  extern __shared__ float lds[];
  float* scores = lds;                         // [16][SSTRIDE]
  float* accum  = lds + 16 * SSTRIDE;          // [16][SSTRIDE] head-mean accum
  float* red    = lds + 2 * 16 * SSTRIDE;      // [16][16]
  float* rowmax = red + 256;                   // [16]
  float* rowsum = rowmax + 16;                 // [16]

  const int b     = blockIdx.x >> 7;           // / (S/16)
  const int qt    = blockIdx.x & 127;
  const int qbase = qt * 16;

  const int wave    = threadIdx.x >> 5;
  const int lane    = threadIdx.x & 31;
  const int col     = lane & 15;
  const int rowbase = (lane >> 4) << 3;

  const int r16 = threadIdx.x >> 4;            // softmax row 0..15
  const int c16 = threadIdx.x & 15;            // softmax col group

  const float inv16 = 1.0f / 16.0f;

  for (int h = 0; h < NH; ++h) {
    const _Float16* qh = qbuf + (((size_t)b * NH + h) * S_LEN + qbase) * HD;
    const _Float16* kh = kbuf + (((size_t)b * NH + h) * S_LEN) * HD;

    const v16h a0 = load_a_frag(qh,      HD);  // k = 0..31
    const v16h a1 = load_a_frag(qh + 32, HD);  // k = 32..63

    for (int it = 0; it < 16; ++it) {
      const int keybase = (it * 8 + wave) * 16;
      const _Float16* kp = kh + (size_t)keybase * HD;
      if (it < 15)  // hint next tile into cache
        __builtin_prefetch(kh + (size_t)(keybase + 128) * HD, 0, 1);
      const v16h b0 = load_b_frag(kp,      HD);
      const v16h b1 = load_b_frag(kp + 32, HD);
      v8f c = {};
      c = __builtin_amdgcn_wmma_f32_16x16x32_f16(false, a0, false, b0,
                                                 (short)0, c, false, false);
      c = __builtin_amdgcn_wmma_f32_16x16x32_f16(false, a1, false, b1,
                                                 (short)0, c, false, false);
#pragma unroll
      for (int r = 0; r < 8; ++r)
        scores[(rowbase + r) * SSTRIDE + keybase + col] = c[r];
    }
    __syncthreads();

    // ---- row max ----
    float m = -1e30f;
    for (int c2 = c16; c2 < S_LEN; c2 += 16)
      m = fmaxf(m, scores[r16 * SSTRIDE + c2]);
    red[r16 * 16 + c16] = m;
    __syncthreads();
    if (c16 == 0) {
      float mm = red[r16 * 16];
#pragma unroll
      for (int i = 1; i < 16; ++i) mm = fmaxf(mm, red[r16 * 16 + i]);
      rowmax[r16] = mm;
    }
    __syncthreads();
    const float rm = rowmax[r16];

    // ---- exp + row sum (exp written back to LDS) ----
    float s = 0.0f;
    for (int c2 = c16; c2 < S_LEN; c2 += 16) {
      const float e = __expf(scores[r16 * SSTRIDE + c2] - rm);
      scores[r16 * SSTRIDE + c2] = e;
      s += e;
    }
    red[r16 * 16 + c16] = s;
    __syncthreads();
    if (c16 == 0) {
      float ss = 0.0f;
#pragma unroll
      for (int i = 0; i < 16; ++i) ss += red[r16 * 16 + i];
      rowsum[r16] = ss;
    }
    __syncthreads();
    const float inv = inv16 / rowsum[r16];

    // ---- accumulate head mean in LDS; write global out once on last head ----
    if (h == 0) {
      for (int c2 = c16; c2 < S_LEN; c2 += 16)
        accum[r16 * SSTRIDE + c2] = scores[r16 * SSTRIDE + c2] * inv;
    } else if (h < NH - 1) {
      for (int c2 = c16; c2 < S_LEN; c2 += 16)
        accum[r16 * SSTRIDE + c2] += scores[r16 * SSTRIDE + c2] * inv;
    } else {
      float* orow = out + ((size_t)b * S_LEN + qbase + r16) * S_LEN;
      for (int c2 = c16; c2 < S_LEN; c2 += 16)
        orow[c2] = accum[r16 * SSTRIDE + c2] + scores[r16 * SSTRIDE + c2] * inv;
    }
    __syncthreads();   // LDS reused next head
  }
}

// ---------------------------------------------------------------------------
extern "C" void kernel_launch(void* const* d_in, const int* in_sizes, int n_in,
                              void* d_out, int out_size, void* d_ws, size_t ws_size,
                              hipStream_t stream) {
  const float* query = (const float*)d_in[0];   // [B,S,D]
  const float* key   = (const float*)d_in[1];   // [B,S,D]
  const float* Wq    = (const float*)d_in[2];   // [D,D]
  const float* bq    = (const float*)d_in[3];   // [D]
  const float* Wk    = (const float*)d_in[4];   // [D,D]
  const float* bk    = (const float*)d_in[5];   // [D]
  float* out = (float*)d_out;                   // [B,S,S]

  const int XN = B_SZ * S_LEN * D_DIM;          // 4 Mi elements
  const int WN = D_DIM * D_DIM;                 // 1 Mi elements

  char* ws = (char*)d_ws;
  _Float16* Xq16 = (_Float16*)ws;  ws += (size_t)XN * 2;
  _Float16* Xk16 = (_Float16*)ws;  ws += (size_t)XN * 2;
  _Float16* Wq16 = (_Float16*)ws;  ws += (size_t)WN * 2;
  _Float16* Wk16 = (_Float16*)ws;  ws += (size_t)WN * 2;
  _Float16* qbuf = (_Float16*)ws;  ws += (size_t)XN * 2;
  _Float16* kbuf = (_Float16*)ws;

  // 1) f32 -> f16 conversions
  cvt_kernel<<<dim3(XN / 1024), 256, 0, stream>>>(query, Xq16, XN);
  cvt_kernel<<<dim3(XN / 1024), 256, 0, stream>>>(key,   Xk16, XN);
  cvt_kernel<<<dim3(WN / 1024), 256, 0, stream>>>(Wq,    Wq16, WN);
  cvt_kernel<<<dim3(WN / 1024), 256, 0, stream>>>(Wk,    Wk16, WN);

  // 2) projections (Q gets the 1/sqrt(HD) scale folded in)
  const float qscale = 0.125f;   // 1/sqrt(64)
  proj_kernel<<<dim3((B_SZ * S_LEN) / 16, D_DIM / 128), 256, 0, stream>>>(
      Xq16, Wq16, bq, qbuf, qscale);
  proj_kernel<<<dim3((B_SZ * S_LEN) / 16, D_DIM / 128), 256, 0, stream>>>(
      Xk16, Wk16, bk, kbuf, 1.0f);

  // 3) scores + softmax + head mean (LDS-resident accumulator)
  const size_t smem = (size_t)(2 * 16 * SSTRIDE + 256 + 16 + 16) * sizeof(float);
  attn_kernel<<<dim3(B_SZ * (S_LEN / 16)), 256, smem, stream>>>(qbuf, kbuf, out);
}
